// PerVarDecoder_46505905881353
// MI455X (gfx1250) — compile-verified, matrix operand order checked by reading, not run on hardware
//
#include <hip/hip_runtime.h>
#include <hip/hip_bf16.h>

// Problem constants (from reference): D=512, L=256, B=512, NL=2
constexpr int kD = 512;
constexpr int kL = 256;
constexpr int kB = 512;

// LDS A-tile: 32 rows x 512 K, padded +8 bf16 (16B) per row to spread banks.
constexpr int kLdsStride = 520;

typedef __attribute__((ext_vector_type(16))) __bf16 v16bf;
typedef __attribute__((ext_vector_type(8)))  __bf16 v8bf;
typedef __attribute__((ext_vector_type(8)))  float  v8f;
typedef __attribute__((ext_vector_type(4)))  unsigned int v4u;
typedef __attribute__((ext_vector_type(8)))  int v8i;
typedef __attribute__((ext_vector_type(4)))  int v4i;

#define WMMA_BF16(a, b, c) \
  __builtin_amdgcn_wmma_f32_16x16x32_bf16(false, (a), false, (b), (short)0, (c), false, false)

// ---------------------------------------------------------------------------
// Fast activations: CDNA5 has native V_TANH_F32 (TRANS op, co-executes with
// WMMA). Guarded so host pass / other toolchains fall back to libm.
// ---------------------------------------------------------------------------
__device__ __forceinline__ float tanh_fast(float x) {
#if __has_builtin(__builtin_amdgcn_tanhf)
  return __builtin_amdgcn_tanhf(x);
#elif __has_builtin(__builtin_amdgcn_tanh_f32)
  return __builtin_amdgcn_tanh_f32(x);
#else
  return ::tanhf(x);
#endif
}
__device__ __forceinline__ float sigmoid_fast(float x) {
  return 0.5f * tanh_fast(0.5f * x) + 0.5f;  // branchless, 1 TRANS + 2 VALU
}

// ---------------------------------------------------------------------------
// Fragment loader for 16-bit A (16x32 MxK) / B (32x16 KxN fed as W rows).
// ISA 7.12.2 layout: lanes 0-15 hold K 0..7 & 16..23, lanes 16-31 hold
// K 8..15 & 24..31 -> two 16-byte loads per lane. Works for both global
// pointers (global_load_b128) and LDS pointers (ds_load_b128 after inlining).
// ---------------------------------------------------------------------------
__device__ __forceinline__ v16bf load_frag(const __bf16* base, int ld) {
  const int lane = threadIdx.x & 31;
  const __bf16* p = base + (size_t)(lane & 15) * ld + ((lane >> 4) << 3);
  v8bf lo = *(const v8bf*)(p);
  v8bf hi = *(const v8bf*)(p + 16);
  v16bf f;
#pragma unroll
  for (int i = 0; i < 8; ++i) { f[i] = lo[i]; f[i + 8] = hi[i]; }
  return f;
}

// ---------------------------------------------------------------------------
// Stage a 32 x 512 bf16 activation tile into LDS.
// Preferred path: Tensor Data Mover (TENSOR_LOAD_TO_LDS) issued by wave 0,
// with a 2D D# descriptor (data_size=2B, tensor 512x512 elements, tile
// 512(K) x 32(rows), row stride 512, LDS padding 4 dwords per 256 dwords so
// the padded LDS row stride is kLdsStride=520 elements).
// Fallback: cooperative vectorized copy by all 256 threads.
// Caller must __syncthreads() after.
// ---------------------------------------------------------------------------
#if defined(__AMDGCN__) && __has_builtin(__builtin_amdgcn_tensor_load_to_lds)
#define USE_TDM 1
#else
#define USE_TDM 0
#endif

__device__ __forceinline__ void stage_a_tile(__bf16* smA, const __bf16* gsrc) {
#if USE_TDM
  if ((threadIdx.x >> 5) == 0) {
    const unsigned long long ga = (unsigned long long)(size_t)gsrc;
    v4u g0;
    g0[0] = 1u;                                   // count = 1 valid descriptor
    g0[1] = (unsigned)(size_t)smA;                // LDS byte offset (low 32 bits)
    g0[2] = (unsigned)ga;                         // global_addr[31:0]
    g0[3] = (unsigned)((ga >> 32) & 0x01FFFFFFu)  // global_addr[56:32]
            | 0x80000000u;                        // type = 2 ("image")
    v8i g1;
    g1[0] = (int)((1u << 16)      // data_size = 1 -> 2 bytes
                | (1u << 20)      // pad_enable
                | (7u << 22)      // pad_interval: every 256 dwords (= one row)
                | (3u << 25));    // pad_amount: 4 dwords (16B)
    g1[1] = (int)(512u << 16);    // tensor_dim0 = 512 (K elements per row)
    g1[2] = (int)(512u << 16);    // tensor_dim1 = 512 (rows in tensor)
    g1[3] = (int)(512u << 16);    // tile_dim0   = 512
    g1[4] = 32;                   // tile_dim1   = 32 rows
    g1[5] = 512;                  // tensor_dim0_stride = 512 elements
    g1[6] = 0;
    g1[7] = 0;
    v4i g2 = {0, 0, 0, 0};        // 2D tensor: groups 2/3 unused
    v4i g3 = {0, 0, 0, 0};
    v8i g4 = {0, 0, 0, 0, 0, 0, 0, 0};  // extra group slot (6-arg builtin form)
    __builtin_amdgcn_tensor_load_to_lds(g0, g1, g2, g3, g4, 0);
#if __has_builtin(__builtin_amdgcn_s_wait_tensorcnt)
    __builtin_amdgcn_s_wait_tensorcnt(0);
#endif
  }
#else
  for (int c = threadIdx.x; c < 32 * (kD / 8); c += 256) {
    const int r = c >> 6, cc = c & 63;
    *(v8bf*)&smA[r * kLdsStride + cc * 8] =
        *(const v8bf*)&gsrc[(size_t)r * kD + cc * 8];
  }
#endif
}

// ---------------------------------------------------------------------------
// f32 -> bf16 conversion (once per launch for inputs/weights).
// ---------------------------------------------------------------------------
__global__ void f32_to_bf16_kernel(const float* __restrict__ in,
                                   __bf16* __restrict__ out, int n) {
  int i = blockIdx.x * blockDim.x + threadIdx.x;
  if (i < n) out[i] = (__bf16)in[i];
}

// ---------------------------------------------------------------------------
// Init GEMM: out = tanh(A(512x512) @ W^T(512x1024) + bias), columns 0..511 ->
// layer-0 state buffer, 512..1023 -> layer-1. Writes bf16 (h) or f32 (c).
// Block: 8 waves = 64(M) x 32(N) tile. Grid: (1024/32, 512/64).
// ---------------------------------------------------------------------------
__global__ __launch_bounds__(256)
void init_gemm_kernel(const __bf16* __restrict__ A,      // [B, D] bf16
                      const __bf16* __restrict__ W,      // [2D, D] bf16, K-contig rows
                      const float*  __restrict__ bias,   // [2D]
                      __bf16* __restrict__ out_bf_l0, __bf16* __restrict__ out_bf_l1,
                      float*  __restrict__ out_f_l0,  float*  __restrict__ out_f_l1) {
  const int wave = threadIdx.x >> 5;
  const int lane = threadIdx.x & 31;
  const int wm = wave & 3;
  const int wn = wave >> 2;
  const int mbase = blockIdx.y * 64 + wm * 16;
  const int nbase = blockIdx.x * 32 + wn * 16;

  v8f acc = {};
  const __bf16* Ap = A + (size_t)mbase * kD;
  const __bf16* Wp = W + (size_t)nbase * kD;
#pragma unroll 4
  for (int k = 0; k < kD; k += 32) {
    v16bf a = load_frag(Ap + k, kD);
    v16bf b = load_frag(Wp + k, kD);
    acc = WMMA_BF16(a, b, acc);
  }

  const int n     = nbase + (lane & 15);
  const int layer = n >> 9;
  const int col   = n & (kD - 1);
  const float bn  = bias[n];
#pragma unroll
  for (int r = 0; r < 8; ++r) {
    const int row = mbase + ((lane >> 4) << 3) + r;
    const float v = tanh_fast(acc[r] + bn);
    const size_t idx = (size_t)row * kD + col;
    if (out_bf_l0) {
      if (layer == 0) out_bf_l0[idx] = (__bf16)v; else out_bf_l1[idx] = (__bf16)v;
    } else {
      if (layer == 0) out_f_l0[idx] = v; else out_f_l1[idx] = v;
    }
  }
}

// ---------------------------------------------------------------------------
// Fused LSTM step: gates = [x_in @ Wih^T +] h_in @ Whh^T + b_ih + b_hh, then
// the cell update, all in registers. Activation tile staged into LDS by the
// TDM; weights stream from L2 via global_load_b128.
// Each wave: one 16x16 tile of each gate chunk (i,f,g,o) -> 4 accumulators,
// 4 WMMAs per 32-wide K chunk. Block: 8 waves as 2(M) x 4(N) = 32 batch x
// 64 gate-cols, so 4 waves share every LDS A fragment.
// Grid: (512/64, 512/32) = (8, 16). c updated in place; h ping-pongs.
// ---------------------------------------------------------------------------
__global__ __launch_bounds__(256)
void lstm_step_kernel(const __bf16* __restrict__ h_in,   // [B, D]
                      const __bf16* __restrict__ x_in,   // [B, D] or nullptr (layer 0)
                      const __bf16* __restrict__ Whh,    // [4D, D]
                      const __bf16* __restrict__ Wih,    // [4D, D] or nullptr
                      const float*  __restrict__ b_ih,   // [4D]
                      const float*  __restrict__ b_hh,   // [4D]
                      float*        __restrict__ c_state,// [B, D] in-place
                      __bf16*       __restrict__ h_out)  // [B, D]
{
  __shared__ __bf16 smA[32 * kLdsStride];   // 33,280 bytes

  const int wave = threadIdx.x >> 5;
  const int lane = threadIdx.x & 31;
  const int wm = wave & 1;                  // 2 waves along batch
  const int wn = wave >> 1;                 // 4 waves along gate cols
  const int mblock = blockIdx.y * 32;
  const int mbase  = mblock + wm * 16;
  const int jbase  = blockIdx.x * 64 + wn * 16;

  v8f ai = {}, af = {}, ag = {}, ao = {};

  const __bf16* W0 = Whh + (size_t)(0 * kD + jbase) * kD;
  const __bf16* W1 = Whh + (size_t)(1 * kD + jbase) * kD;
  const __bf16* W2 = Whh + (size_t)(2 * kD + jbase) * kD;
  const __bf16* W3 = Whh + (size_t)(3 * kD + jbase) * kD;

  stage_a_tile(smA, h_in + (size_t)mblock * kD);   // TDM -> LDS
  __syncthreads();

  const __bf16* As = &smA[wm * 16 * kLdsStride];
  for (int k = 0; k < kD; k += 32) {
    if (k + 32 < kD) {   // global_prefetch_b8 of next weight chunk
      __builtin_prefetch(W0 + k + 32, 0, 0);
      __builtin_prefetch(W2 + k + 32, 0, 0);
    }
    v16bf a  = load_frag(As + k, kLdsStride);      // ds_load_b128
    v16bf b0 = load_frag(W0 + k, kD);
    v16bf b1 = load_frag(W1 + k, kD);
    v16bf b2 = load_frag(W2 + k, kD);
    v16bf b3 = load_frag(W3 + k, kD);
    ai = WMMA_BF16(a, b0, ai);
    af = WMMA_BF16(a, b1, af);
    ag = WMMA_BF16(a, b2, ag);
    ao = WMMA_BF16(a, b3, ao);
  }

  if (x_in != nullptr) {   // layer-1: fused y0[t] @ Wih^T segment (K += 512)
    const __bf16* V0 = Wih + (size_t)(0 * kD + jbase) * kD;
    const __bf16* V1 = Wih + (size_t)(1 * kD + jbase) * kD;
    const __bf16* V2 = Wih + (size_t)(2 * kD + jbase) * kD;
    const __bf16* V3 = Wih + (size_t)(3 * kD + jbase) * kD;

    __syncthreads();                                // all reads of smA done
    stage_a_tile(smA, x_in + (size_t)mblock * kD);  // restage with x tile
    __syncthreads();

    for (int k = 0; k < kD; k += 32) {
      if (k + 32 < kD) {
        __builtin_prefetch(V0 + k + 32, 0, 0);
        __builtin_prefetch(V2 + k + 32, 0, 0);
      }
      v16bf a  = load_frag(As + k, kLdsStride);
      v16bf b0 = load_frag(V0 + k, kD);
      v16bf b1 = load_frag(V1 + k, kD);
      v16bf b2 = load_frag(V2 + k, kD);
      v16bf b3 = load_frag(V3 + k, kD);
      ai = WMMA_BF16(a, b0, ai);
      af = WMMA_BF16(a, b1, af);
      ag = WMMA_BF16(a, b2, ag);
      ao = WMMA_BF16(a, b3, ao);
    }
  }

  // Cell update. C/D layout: VGPR r of lane l holds M = (l>=16)*8 + r, N = l&15.
  const int j = jbase + (lane & 15);
  const float bi  = b_ih[j]          + b_hh[j];
  const float bff = b_ih[kD + j]     + b_hh[kD + j];
  const float bg  = b_ih[2 * kD + j] + b_hh[2 * kD + j];
  const float bo  = b_ih[3 * kD + j] + b_hh[3 * kD + j];
#pragma unroll
  for (int r = 0; r < 8; ++r) {
    const int row = mbase + ((lane >> 4) << 3) + r;
    const size_t idx = (size_t)row * kD + j;
    const float ig = sigmoid_fast(ai[r] + bi);
    const float fg = sigmoid_fast(af[r] + bff);
    const float gg = tanh_fast(ag[r] + bg);
    const float og = sigmoid_fast(ao[r] + bo);
    const float cn = fg * c_state[idx] + ig * gg;
    c_state[idx] = cn;
    h_out[idx] = (__bf16)(og * tanh_fast(cn));
  }
}

// ---------------------------------------------------------------------------
// Output projection for one timestep: O[b] = y1_t[b,:] . out_w + out_b,
// scattered into recon[(B, L-1)] / pred[(B,)] layout of d_out.
// ---------------------------------------------------------------------------
__global__ void out_proj_kernel(const __bf16* __restrict__ y,     // [B, D]
                                const float*  __restrict__ out_w, // [D]
                                const float*  __restrict__ out_b, // [1]
                                float* __restrict__ out, int t) {
  const int b = blockIdx.x * blockDim.x + threadIdx.x;
  if (b >= kB) return;
  const v8bf* yp = (const v8bf*)(y + (size_t)b * kD);
  float s = out_b[0];
#pragma unroll 4
  for (int i = 0; i < kD / 8; ++i) {
    v8bf v = yp[i];
#pragma unroll
    for (int jj = 0; jj < 8; ++jj) s += (float)v[jj] * out_w[i * 8 + jj];
  }
  if (t < kL - 1) out[(size_t)b * (kL - 1) + t] = s;
  else            out[(size_t)kB * (kL - 1) + b] = s;
}

// ---------------------------------------------------------------------------
// Launch sequence:
//   converts -> init h/c GEMMs -> 256 x (layer0 step, layer1 step, out proj)
// ---------------------------------------------------------------------------
extern "C" void kernel_launch(void* const* d_in, const int* in_sizes, int n_in,
                              void* d_out, int out_size, void* d_ws, size_t ws_size,
                              hipStream_t stream) {
  (void)in_sizes; (void)n_in; (void)out_size; (void)ws_size;

  const float* c_star   = (const float*)d_in[0];
  const float* init_h_w = (const float*)d_in[1];
  const float* init_h_b = (const float*)d_in[2];
  const float* init_c_w = (const float*)d_in[3];
  const float* init_c_b = (const float*)d_in[4];
  // d_in[5] = W_ih0 : unused by the reference math (layer-0 input term is bias-only)
  const float* W_hh0    = (const float*)d_in[6];
  const float* b_ih0    = (const float*)d_in[7];
  const float* b_hh0    = (const float*)d_in[8];
  const float* W_ih1    = (const float*)d_in[9];
  const float* W_hh1    = (const float*)d_in[10];
  const float* b_ih1    = (const float*)d_in[11];
  const float* b_hh1    = (const float*)d_in[12];
  const float* out_w    = (const float*)d_in[13];
  const float* out_b    = (const float*)d_in[14];
  float* out = (float*)d_out;

  // Workspace carve-out (~12.5 MB), 256B-aligned slices.
  char* p = (char*)d_ws;
  auto alloc = [&](size_t bytes) {
    char* r = p;
    p += (bytes + 255) & ~(size_t)255;
    return r;
  };
  const size_t nBD  = (size_t)kB * kD;          // 262144
  const size_t n2DD = (size_t)2 * kD * kD;      // 524288
  const size_t n4DD = (size_t)4 * kD * kD;      // 1048576

  __bf16* cs_bf   = (__bf16*)alloc(nBD * 2);
  __bf16* ihw_bf  = (__bf16*)alloc(n2DD * 2);
  __bf16* icw_bf  = (__bf16*)alloc(n2DD * 2);
  __bf16* whh0_bf = (__bf16*)alloc(n4DD * 2);
  __bf16* wih1_bf = (__bf16*)alloc(n4DD * 2);
  __bf16* whh1_bf = (__bf16*)alloc(n4DD * 2);
  __bf16* h0a = (__bf16*)alloc(nBD * 2);
  __bf16* h0b = (__bf16*)alloc(nBD * 2);
  __bf16* h1a = (__bf16*)alloc(nBD * 2);
  __bf16* h1b = (__bf16*)alloc(nBD * 2);
  float*  c0  = (float*)alloc(nBD * 4);
  float*  c1  = (float*)alloc(nBD * 4);

  auto cvt = [&](const float* src, __bf16* dst, size_t n) {
    f32_to_bf16_kernel<<<dim3((unsigned)((n + 255) / 256)), dim3(256), 0, stream>>>(src, dst, (int)n);
  };
  cvt(c_star,   cs_bf,   nBD);
  cvt(init_h_w, ihw_bf,  n2DD);
  cvt(init_c_w, icw_bf,  n2DD);
  cvt(W_hh0,    whh0_bf, n4DD);
  cvt(W_ih1,    wih1_bf, n4DD);
  cvt(W_hh1,    whh1_bf, n4DD);

  // Init states: h0/c0 for both layers (N = 2*D = 1024 columns split by layer)
  dim3 gi(2 * kD / 32, kB / 64);  // (32, 8)
  init_gemm_kernel<<<gi, 256, 0, stream>>>(cs_bf, ihw_bf, init_h_b,
                                           h0a, h1a, nullptr, nullptr);
  init_gemm_kernel<<<gi, 256, 0, stream>>>(cs_bf, icw_bf, init_c_b,
                                           nullptr, nullptr, c0, c1);

  __bf16* h0_in = h0a; __bf16* h0_out = h0b;
  __bf16* h1_in = h1a; __bf16* h1_out = h1b;
  dim3 gs(kD / 64, kB / 32);      // (8, 16)
  for (int t = 0; t < kL; ++t) {
    // Layer 0: input term is (b_ih0 + b_hh0) broadcast -> bias-only.
    lstm_step_kernel<<<gs, 256, 0, stream>>>(h0_in, nullptr, whh0_bf, nullptr,
                                             b_ih0, b_hh0, c0, h0_out);
    // Layer 1: fused y0[t] @ W_ih1^T segment + recurrent segment.
    lstm_step_kernel<<<gs, 256, 0, stream>>>(h1_in, h0_out, whh1_bf, wih1_bf,
                                             b_ih1, b_hh1, c1, h1_out);
    out_proj_kernel<<<dim3(2), dim3(256), 0, stream>>>(h1_out, out_w, out_b, out, t);
    __bf16* tmp;
    tmp = h0_in; h0_in = h0_out; h0_out = tmp;
    tmp = h1_in; h1_in = h1_out; h1_out = tmp;
  }
}
